// MultiHeadAttention_60816736911814
// MI455X (gfx1250) — compile-verified, matrix operand order checked by reading, not run on hardware
//
#include <hip/hip_runtime.h>
#include <hip/hip_bf16.h>

#define D_MODEL 1280
#define NUM_HEADS 16
#define D_K 80
#define GROUPS 5
#define BATCH 4
#define SEQ 1024
#define NBLK (SEQ / 32)

typedef __attribute__((ext_vector_type(16))) __bf16 v16bf;
typedef __attribute__((ext_vector_type(8)))  __bf16 v8bf;
typedef __attribute__((ext_vector_type(8)))  float  v8f;
typedef __attribute__((ext_vector_type(2)))  int    v2i;
typedef __attribute__((ext_vector_type(4)))  int    v4i;

union Frag { v16bf v; v8bf h[2]; __bf16 e[16]; };
union Acc  { v8f v; float f[8]; };

#if __has_builtin(__builtin_amdgcn_global_load_async_to_lds_b64) && \
    __has_builtin(__builtin_amdgcn_global_load_async_to_lds_b128)
#define USE_ASYNC 1
#else
#define USE_ASYNC 0
#endif

__device__ __forceinline__ void async_b64(const void* g, void* l) {
#if USE_ASYNC
  __builtin_amdgcn_global_load_async_to_lds_b64(
      (__attribute__((address_space(1))) v2i*)g,
      (__attribute__((address_space(3))) v2i*)l, 0, 0);
#endif
}
__device__ __forceinline__ void async_b128(const void* g, void* l) {
#if USE_ASYNC
  __builtin_amdgcn_global_load_async_to_lds_b128(
      (__attribute__((address_space(1))) v4i*)g,
      (__attribute__((address_space(3))) v4i*)l, 0, 0);
#endif
}
template <int N> __device__ __forceinline__ void wait_async() {
#if USE_ASYNC
#if __has_builtin(__builtin_amdgcn_s_wait_asynccnt)
  __builtin_amdgcn_s_wait_asynccnt(N);
#else
  asm volatile("s_wait_asynccnt %0" ::"n"(N) : "memory");
#endif
#endif
}

// round-to-nearest-even f32 -> bf16
__device__ __forceinline__ __bf16 f2bf(float x) {
  unsigned u = __builtin_bit_cast(unsigned, x);
  unsigned r = u + 0x7FFFu + ((u >> 16) & 1u);
  unsigned short s = (unsigned short)(r >> 16);
  return __builtin_bit_cast(__bf16, s);
}

// Load one 16x32 bf16 operand fragment (A: rows=M, B: rows=N) from a
// row-major tile.  Lane layout per CDNA5 ISA (16-bit A/B matrices):
//   lanes 0-15 : row = lane,    K = k0+{0..7, 16..23}
//   lanes 16-31: row = lane-16, K = k0+{8..15, 24..31}
// tail=true zeroes the upper 8 K-slots (row length 80, k0==64 padding).
__device__ __forceinline__ v16bf ld_frag(const __bf16* base, int row0,
                                         int rstride, int k0, bool tail) {
  int lane = threadIdx.x & 31;
  int half = lane >> 4;
  int r = lane & 15;
  const __bf16* p = base + (size_t)(row0 + r) * rstride + k0 + half * 8;
  Frag f;
  f.h[0] = *(const v8bf*)p;
  if (tail) {
#pragma unroll
    for (int i = 0; i < 8; ++i) f.e[8 + i] = f2bf(0.f);
  } else {
    f.h[1] = *(const v8bf*)(p + 16);
  }
  return f.v;
}

__device__ __forceinline__ v8f wmma_bf16(v16bf a, v16bf b, v8f c) {
  return __builtin_amdgcn_wmma_f32_16x16x32_bf16(false, a, false, b,
                                                 (short)0, c, false, false);
}

// ---------------------------------------------------------------------------
// Kernel 0: Wo f32 -> bf16 (so the output-proj kernel can async-stage it)
// ---------------------------------------------------------------------------
__global__ __launch_bounds__(256)
void cvt_wo_kernel(const float* __restrict__ Wo, __bf16* __restrict__ Wob) {
  size_t i = (size_t)blockIdx.x * 256 + threadIdx.x;
  Wob[i] = f2bf(Wo[i]);
}

// ---------------------------------------------------------------------------
// Kernel 1: channel shuffle + per-head Q/K/V projections.
// Q/K computed with A=W (M=feature) so each lane's 8 accum values are
// contiguous in [S,80] -> packed b128 stores.  V keeps A=x (M=seq) so the
// transposed V^T[80,S] store is contiguous.  Q pre-scaled by 1/sqrt(80).
// ---------------------------------------------------------------------------
__global__ __launch_bounds__(256)
void proj_kernel(const float* __restrict__ src,
                 const float* __restrict__ Wq, const float* __restrict__ bq,
                 const float* __restrict__ Wk, const float* __restrict__ bk,
                 const float* __restrict__ Wv, const float* __restrict__ bv,
                 __bf16* __restrict__ Qs, __bf16* __restrict__ Ks,
                 __bf16* __restrict__ Vt) {
  __shared__ __bf16 xs[128][96] __attribute__((aligned(16)));
  __shared__ __bf16 wsm[80][96] __attribute__((aligned(16)));

  int blk  = blockIdx.x;
  int sblk = blk % (SEQ / 128);
  int bh   = blk / (SEQ / 128);
  int h    = bh % NUM_HEADS;
  int tid  = threadIdx.x;
  int wave = tid >> 5;
  int lane = tid & 31;
  int half = lane >> 4;
  int lrow = lane & 15;

  const float* srcb =
      src + ((size_t)(bh / NUM_HEADS) * SEQ + (size_t)sblk * 128) * D_MODEL;
  for (int i = tid; i < 128 * 96; i += 256) {
    int rr = i / 96, cc = i % 96;
    float v = 0.f;
    if (cc < D_K) {
      int dp = h * D_K + cc;  // shuffled channel
      int sc = (dp % GROUPS) * (D_MODEL / GROUPS) + dp / GROUPS;
      v = srcb[(size_t)rr * D_MODEL + sc];
    }
    xs[rr][cc] = f2bf(v);
  }
  __syncthreads();

  int r0 = wave * 16;
  v16bf a0 = ld_frag(&xs[0][0], r0, 96, 0,  false);
  v16bf a1 = ld_frag(&xs[0][0], r0, 96, 32, false);
  v16bf a2 = ld_frag(&xs[0][0], r0, 96, 64, false);

  const float scale = 0.11180339887498949f;  // 1/sqrt(80)

#pragma unroll
  for (int m = 0; m < 3; ++m) {
    const float* W  = (m == 0 ? Wq : (m == 1 ? Wk : Wv)) + (size_t)h * D_K * D_K;
    const float* Bb = (m == 0 ? bq : (m == 1 ? bk : bv));
    __syncthreads();
    for (int i = tid; i < 80 * 96; i += 256) {
      int rr = i / 96, cc = i % 96;
      float v = (cc < D_K) ? W[rr * D_K + cc] : 0.f;
      wsm[rr][cc] = f2bf(v);
    }
    __syncthreads();

#pragma unroll
    for (int n = 0; n < 5; ++n) {
      v16bf w0 = ld_frag(&wsm[0][0], n * 16, 96, 0,  false);
      v16bf w1 = ld_frag(&wsm[0][0], n * 16, 96, 32, false);
      v16bf w2 = ld_frag(&wsm[0][0], n * 16, 96, 64, false);
      Acc c;
#pragma unroll
      for (int i = 0; i < 8; ++i) c.f[i] = 0.f;
      if (m < 2) {  // M = feature, N = seq row
        c.v = wmma_bf16(w0, a0, c.v);
        c.v = wmma_bf16(w1, a1, c.v);
        c.v = wmma_bf16(w2, a2, c.v);
        int srow = sblk * 128 + r0 + lrow;  // N
        v8bf pv;
#pragma unroll
        for (int r = 0; r < 8; ++r) {
          float bi = Bb[h * D_K + n * 16 + half * 8 + r];
          float val = c.f[r] + bi;
          pv[r] = f2bf(m == 0 ? val * scale : val);
        }
        __bf16* dst = (m == 0 ? Qs : Ks) +
                      ((size_t)bh * SEQ + srow) * D_K + n * 16 + half * 8;
        *(v8bf*)dst = pv;
      } else {  // V: M = seq row, N = feature -> V^T store contiguous in s
        c.v = wmma_bf16(a0, w0, c.v);
        c.v = wmma_bf16(a1, w1, c.v);
        c.v = wmma_bf16(a2, w2, c.v);
        int col = n * 16 + lrow;  // N = feature e
        float bi = Bb[h * D_K + col];
        v8bf pv;
#pragma unroll
        for (int r = 0; r < 8; ++r) pv[r] = f2bf(c.f[r] + bi);
        __bf16* dst = Vt + ((size_t)bh * D_K + col) * SEQ + sblk * 128 + r0 +
                      half * 8;
        *(v8bf*)dst = pv;
      }
    }
  }
}

// ---------------------------------------------------------------------------
// Kernel 2: flash attention, transposed (St = K*Q^T, O^T = V^T * P^T),
// with double-buffered async LDS staging of K / V^T blocks (ASYNCcnt).
// Each wave issues exactly 5 async b64 loads per buffer.
// ---------------------------------------------------------------------------
__device__ __forceinline__ void stage_kv(int buf, int kb, int tid,
                                         const __bf16* Kbase,
                                         const __bf16* Vbase,
                                         __bf16 (*kls)[32][80],
                                         __bf16 (*vls)[80][32]) {
#if USE_ASYNC
#pragma unroll
  for (int t = 0; t < 5; ++t) {  // 640 K-chunks + 640 V-chunks of 8 bytes
    int j = tid + t * 256;
    const __bf16* g;
    __bf16* l;
    if (j < 640) {
      int rr = j / 20, cc = (j % 20) * 4;
      g = Kbase + (size_t)(kb * 32 + rr) * D_K + cc;
      l = &kls[buf][rr][cc];
    } else {
      int j2 = j - 640;
      int rr = j2 / 8, cc = (j2 % 8) * 4;
      g = Vbase + (size_t)rr * SEQ + kb * 32 + cc;
      l = &vls[buf][rr][cc];
    }
    async_b64(g, l);  // one instruction per t per wave -> ASYNCcnt += 5
  }
#else
  for (int i = tid; i < 320; i += 256) {
    int rr = i / 10, cc = (i % 10) * 8;
    *(v8bf*)&kls[buf][rr][cc] =
        *(const v8bf*)&Kbase[(size_t)(kb * 32 + rr) * D_K + cc];
  }
  for (int i = tid; i < 320; i += 256) {
    int rr = i / 4, cc = (i % 4) * 8;
    *(v8bf*)&vls[buf][rr][cc] =
        *(const v8bf*)&Vbase[(size_t)rr * SEQ + kb * 32 + cc];
  }
#endif
}

__global__ __launch_bounds__(256)
void attn_kernel(const __bf16* __restrict__ Qs, const __bf16* __restrict__ Ks,
                 const __bf16* __restrict__ Vt, __bf16* __restrict__ concat) {
  __shared__ __bf16 kls[2][32][80] __attribute__((aligned(16)));
  __shared__ __bf16 vls[2][80][32] __attribute__((aligned(16)));

  int blk  = blockIdx.x;
  int qblk = blk % (SEQ / 128);
  int bh   = blk / (SEQ / 128);
  int h    = bh % NUM_HEADS;
  int b    = bh / NUM_HEADS;
  int tid  = threadIdx.x;
  int wave = tid >> 5;
  int lane = tid & 31;
  int half = lane >> 4;
  int lrow = lane & 15;

  int q0 = qblk * 128 + wave * 16;
  const __bf16* Qbase = Qs + (size_t)bh * SEQ * D_K;
  const __bf16* Kbase = Ks + (size_t)bh * SEQ * D_K;
  const __bf16* Vbase = Vt + (size_t)bh * D_K * SEQ;

  stage_kv(0, 0, tid, Kbase, Vbase, kls, vls);

  v16bf qf0 = ld_frag(Qbase, q0, D_K, 0,  false);
  v16bf qf1 = ld_frag(Qbase, q0, D_K, 32, false);
  v16bf qf2 = ld_frag(Qbase, q0, D_K, 64, true);

  Acc acc[5];
#pragma unroll
  for (int n = 0; n < 5; ++n)
#pragma unroll
    for (int i = 0; i < 8; ++i) acc[n].f[i] = 0.f;

  float m_run = -1e30f, l_run = 0.f;
  const float LOG2E = 1.4426950408889634f;

  for (int kb = 0; kb < NBLK; ++kb) {
    int cur = kb & 1;
    if (kb + 1 < NBLK) {
      stage_kv(cur ^ 1, kb + 1, tid, Kbase, Vbase, kls, vls);
      wait_async<5>();  // current buffer's 5 loads complete (in-order)
    } else {
      wait_async<0>();
    }
    __syncthreads();

    // preload V^T fragments: ds loads overlap the St WMMA chain + softmax
    v16bf vf[5];
#pragma unroll
    for (int n = 0; n < 5; ++n)
      vf[n] = ld_frag(&vls[cur][0][0], n * 16, 32, 0, false);

    Acc st0, st1;
#pragma unroll
    for (int i = 0; i < 8; ++i) { st0.f[i] = 0.f; st1.f[i] = 0.f; }
    st0.v = wmma_bf16(ld_frag(&kls[cur][0][0], 0,  80, 0,  false), qf0, st0.v);
    st0.v = wmma_bf16(ld_frag(&kls[cur][0][0], 0,  80, 32, false), qf1, st0.v);
    st0.v = wmma_bf16(ld_frag(&kls[cur][0][0], 0,  80, 64, true),  qf2, st0.v);
    st1.v = wmma_bf16(ld_frag(&kls[cur][0][0], 16, 80, 0,  false), qf0, st1.v);
    st1.v = wmma_bf16(ld_frag(&kls[cur][0][0], 16, 80, 32, false), qf1, st1.v);
    st1.v = wmma_bf16(ld_frag(&kls[cur][0][0], 16, 80, 64, true),  qf2, st1.v);

    // online softmax over keys (per query column = per lane, + half swap)
    float mt = -1e30f;
#pragma unroll
    for (int r = 0; r < 8; ++r) {
      mt = fmaxf(mt, st0.f[r]);
      mt = fmaxf(mt, st1.f[r]);
    }
    mt = fmaxf(mt, __shfl_xor(mt, 16));
    float m_new = fmaxf(m_run, mt);
    float alpha = exp2f((m_run - m_new) * LOG2E);

    // C-layout of St == B-operand layout of PV gemm (in-lane pack, no LDS)
    Frag pf;
    float ls = 0.f;
#pragma unroll
    for (int r = 0; r < 8; ++r) {
      float p0 = exp2f((st0.f[r] - m_new) * LOG2E);
      float p1 = exp2f((st1.f[r] - m_new) * LOG2E);
      ls += p0 + p1;
      pf.e[r]     = f2bf(p0);
      pf.e[8 + r] = f2bf(p1);
    }
    ls += __shfl_xor(ls, 16);
    l_run = l_run * alpha + ls;
    m_run = m_new;

#pragma unroll
    for (int n = 0; n < 5; ++n) {
#pragma unroll
      for (int i = 0; i < 8; ++i) acc[n].f[i] *= alpha;
      acc[n].v = wmma_bf16(vf[n], pf.v, acc[n].v);
    }
    __syncthreads();  // all waves done with buf[cur] before it is restaged
  }

  float rl = 1.f / l_run;
  int q = q0 + lrow;  // N = query
#pragma unroll
  for (int n = 0; n < 5; ++n) {
    v8bf pv;
#pragma unroll
    for (int r = 0; r < 8; ++r) pv[r] = f2bf(acc[n].f[r] * rl);
    // lane's 8 values are e = n*16 + half*8 + {0..7}: contiguous
    __bf16* dst = concat + ((size_t)b * SEQ + q) * D_MODEL + h * D_K +
                  n * 16 + half * 8;
    *(v8bf*)dst = pv;
  }
}

// ---------------------------------------------------------------------------
// Kernel 3: output projection  out = concat @ Wo^T + bo  (f32 out), with
// double-buffered async staging (4 b128 async loads per wave per buffer).
// A=Wo (M=out feature) so each lane's accum is contiguous -> v8f stores.
// ---------------------------------------------------------------------------
__device__ __forceinline__ void stage_ow(int buf, int kk, int tid, int cb,
                                         size_t row0,
                                         const __bf16* __restrict__ concat,
                                         const __bf16* __restrict__ Wob,
                                         __bf16 (*xls)[128][32],
                                         __bf16 (*wls)[80][32]) {
#if USE_ASYNC
#pragma unroll
  for (int t = 0; t < 2; ++t) {  // 512 x-chunks of 16B
    int j = tid + t * 256;
    int rr = j / 4, cc = (j % 4) * 8;
    async_b128(&concat[(row0 + rr) * D_MODEL + kk * 32 + cc],
               &xls[buf][rr][cc]);
  }
#pragma unroll
  for (int t = 0; t < 2; ++t) {  // 320 w-chunks, wrapped to 512 (dup benign)
    int j = tid + t * 256;
    int jw = (j >= 320) ? j - 320 : j;
    int rr = jw / 4, cc = (jw % 4) * 8;
    async_b128(&Wob[(size_t)(cb * 80 + rr) * D_MODEL + kk * 32 + cc],
               &wls[buf][rr][cc]);
  }
#else
  for (int i = tid; i < 512; i += 256) {
    int rr = i / 4, cc = (i % 4) * 8;
    *(v8bf*)&xls[buf][rr][cc] =
        *(const v8bf*)&concat[(row0 + rr) * D_MODEL + kk * 32 + cc];
  }
  for (int i = tid; i < 320; i += 256) {
    int rr = i / 4, cc = (i % 4) * 8;
    *(v8bf*)&wls[buf][rr][cc] =
        *(const v8bf*)&Wob[(size_t)(cb * 80 + rr) * D_MODEL + kk * 32 + cc];
  }
#endif
}

__global__ __launch_bounds__(256)
void outproj_kernel(const __bf16* __restrict__ concat,
                    const __bf16* __restrict__ Wob,
                    const float* __restrict__ bo, float* __restrict__ out) {
  __shared__ __bf16 xls[2][128][32] __attribute__((aligned(16)));
  __shared__ __bf16 wls[2][80][32] __attribute__((aligned(16)));

  int blk = blockIdx.x;
  int cb  = blk % (D_MODEL / 80);  // 16 column blocks of 80
  int rb  = blk / (D_MODEL / 80);  // 32 row blocks of 128
  int tid = threadIdx.x;
  int wave = tid >> 5;
  int lane = tid & 31;
  int half = lane >> 4;
  int lrow = lane & 15;
  size_t row0 = (size_t)rb * 128;

  Acc acc[5];
#pragma unroll
  for (int n = 0; n < 5; ++n)
#pragma unroll
    for (int i = 0; i < 8; ++i) acc[n].f[i] = 0.f;

  stage_ow(0, 0, tid, cb, row0, concat, Wob, xls, wls);

  for (int kk = 0; kk < D_MODEL / 32; ++kk) {
    int cur = kk & 1;
    if (kk + 1 < D_MODEL / 32) {
      stage_ow(cur ^ 1, kk + 1, tid, cb, row0, concat, Wob, xls, wls);
      wait_async<4>();
    } else {
      wait_async<0>();
    }
    __syncthreads();

    v16bf xf = ld_frag(&xls[cur][0][0], wave * 16, 32, 0, false);
#pragma unroll
    for (int n = 0; n < 5; ++n)
      acc[n].v = wmma_bf16(ld_frag(&wls[cur][0][0], n * 16, 32, 0, false), xf,
                           acc[n].v);
    __syncthreads();
  }

  // M = out feature, N = seq row: lane's 8 values contiguous in out row
  size_t row = row0 + wave * 16 + lrow;
#pragma unroll
  for (int n = 0; n < 5; ++n) {
    int col0 = cb * 80 + n * 16 + half * 8;
#pragma unroll
    for (int r = 0; r < 8; ++r) acc[n].f[r] += bo[col0 + r];
    *(v8f*)(out + row * D_MODEL + col0) = acc[n].v;
  }
}

// ---------------------------------------------------------------------------
extern "C" void kernel_launch(void* const* d_in, const int* in_sizes, int n_in,
                              void* d_out, int out_size, void* d_ws,
                              size_t ws_size, hipStream_t stream) {
  const float* src = (const float*)d_in[0];
  // d_in[1] = k, d_in[2] = v : ignored by the reference module
  const float* Wq = (const float*)d_in[3];
  const float* bq = (const float*)d_in[4];
  const float* Wk = (const float*)d_in[5];
  const float* bk = (const float*)d_in[6];
  const float* Wv = (const float*)d_in[7];
  const float* bv = (const float*)d_in[8];
  const float* Wo = (const float*)d_in[9];
  const float* bo = (const float*)d_in[10];
  float* out = (float*)d_out;

  size_t nqk = (size_t)BATCH * NUM_HEADS * SEQ * D_K;  // 5,242,880 elems
  __bf16* Qs     = (__bf16*)d_ws;
  __bf16* Ks     = Qs + nqk;
  __bf16* Vt     = Ks + nqk;
  __bf16* concat = Vt + nqk;            // B*S*D_MODEL elems
  __bf16* Wob    = concat + (size_t)BATCH * SEQ * D_MODEL;  // 1280*1280

  cvt_wo_kernel<<<(D_MODEL * D_MODEL) / 256, 256, 0, stream>>>(Wo, Wob);

  int blocks_bh = BATCH * NUM_HEADS * (SEQ / 128);  // 512
  proj_kernel<<<blocks_bh, 256, 0, stream>>>(src, Wq, bq, Wk, bk, Wv, bv, Qs,
                                             Ks, Vt);
  attn_kernel<<<blocks_bh, 256, 0, stream>>>(Qs, Ks, Vt, concat);
  int blocks_o = (BATCH * SEQ / 128) * (D_MODEL / 80);  // 512
  outproj_kernel<<<blocks_o, 256, 0, stream>>>(concat, Wob, bo, out);
}
